// LSTMModel_1898375545335
// MI455X (gfx1250) — matrix operand order repaired and checked by reading, not compiled, not run on hardware
//
#include <hip/hip_runtime.h>
#include <hip/hip_bf16.h>

// ---------------------------------------------------------------------------
// CDNA5 (gfx1250) wave32 WMMA implementation of the peephole-LSTM model.
// GEMMs use v_wmma_f32_16x16x32_bf16 with block tile 64x64:
//   - 4 waves/block, each wave owns a 16x64 slice (4 f32 accumulators)
//   - B tile (64 cols x 32 K) staged cooperatively into LDS once per K-step
//     and shared by all 4 waves (WGP-level reuse; ds_load_b128 fragments)
//   - all 4 B fragments loaded before the 4 back-to-back wmmas (single
//     s_wait_dscnt, back-to-back XDL issue)
// ---------------------------------------------------------------------------

typedef __bf16 bf16;
typedef __attribute__((ext_vector_type(16))) __bf16 v16bf;
typedef __attribute__((ext_vector_type(8)))  __bf16 v8bf;
typedef __attribute__((ext_vector_type(8)))  float  v8f;

#define BATCH   256
#define TSTEPS  128
#define FDIM    784
#define FPAD    800
#define HDIM    1024
#define H4      4096
#define CDIM    10
#define BT      (BATCH * TSTEPS)   // 32768
#define BN_EPS  1e-3f
#define LDB_STR 36                 // LDS row stride (elements), padded vs 32

// ---------------------------------------------------------------------------
// Fragment loaders (wave32; contiguous k = 16*half + s, same order for A and B)
// A 16x32 bf16, row-major source (stride lda elements):
//   lanes 0-15 : M=lane,    V0..7 = K 0..15 (pairs)
//   lanes 16-31: M=lane-16, V0..7 = K 16..31
// ---------------------------------------------------------------------------
__device__ __forceinline__ v16bf load_frag_a(const bf16* __restrict__ A,
                                             long lda, int row0, int k0) {
    const int lane = threadIdx.x & 31;
    const int m    = lane & 15;
    const int half = lane >> 4;
    const bf16* p0 = A + (long)(row0 + m) * lda + (k0 + half * 16);
    const bf16* p1 = p0 + 8;
    v8bf lo = *(const v8bf*)p0;
    v8bf hi = *(const v8bf*)p1;
    v16bf a;
#pragma unroll
    for (int i = 0; i < 8; ++i) { a[i] = lo[i]; a[8 + i] = hi[i]; }
    return a;
}

// B 32x16 fragment from the LDS-staged tile (row = weight column, padded
// stride). lanes 0-15: N=lane, K 0..15; lanes 16-31: N=lane-16, K 16..31.
__device__ __forceinline__ v16bf load_frag_b_lds(const bf16* lbs, int j) {
    const int lane = threadIdx.x & 31;
    const int n    = lane & 15;
    const int half = lane >> 4;
    const bf16* p  = lbs + (16 * j + n) * LDB_STR + half * 16;
    v8bf lo = *(const v8bf*)p;
    v8bf hi = *(const v8bf*)(p + 8);
    v16bf b;
#pragma unroll
    for (int i = 0; i < 8; ++i) { b[i] = lo[i]; b[8 + i] = hi[i]; }
    return b;
}

// B fragment straight from global N-major weights (logits stage).
__device__ __forceinline__ v16bf load_frag_b(const bf16* __restrict__ Wt,
                                             long ldb, int col0, int k0) {
    const int lane = threadIdx.x & 31;
    const int n    = lane & 15;
    const int half = lane >> 4;
    const bf16* p  = Wt + (long)(col0 + n) * ldb + (k0 + half * 16);
    v8bf lo = *(const v8bf*)p;
    v8bf hi = *(const v8bf*)(p + 8);
    v16bf b;
#pragma unroll
    for (int i = 0; i < 8; ++i) { b[i] = lo[i]; b[8 + i] = hi[i]; }
    return b;
}

__device__ __forceinline__ v8f wmma_bf16(v16bf a, v16bf b, v8f c) {
    return __builtin_amdgcn_wmma_f32_16x16x32_bf16(
        /*neg_a=*/false, a, /*neg_b=*/false, b,
        /*c_mod=*/(short)0, c, /*reuse_a=*/false, /*reuse_b=*/false);
}

// Cooperative B-tile stage: 128 threads move 64 cols x 32 K bf16 (4 KB).
__device__ __forceinline__ void stage_b_tile(bf16* lbs, const bf16* __restrict__ Wt,
                                             long ldb, int col0, int k0) {
    const int tid = threadIdx.x;        // 0..127
    const int c   = tid >> 1;           // 0..63
    const int kh  = (tid & 1) * 16;
    const bf16* g = Wt + (long)(col0 + c) * ldb + k0 + kh;
    v8bf t0 = *(const v8bf*)g;
    v8bf t1 = *(const v8bf*)(g + 8);
    *(v8bf*)&lbs[c * LDB_STR + kh]     = t0;
    *(v8bf*)&lbs[c * LDB_STR + kh + 8] = t1;
}

__device__ __forceinline__ float sigm(float x) {
    return 1.0f / (1.0f + __expf(-x));
}

// ---------------------------------------------------------------------------
// One-time data layout conversions (f32 -> bf16, pad / transpose)
// ---------------------------------------------------------------------------
__global__ void pad_cvt_bf16(const float* __restrict__ src, bf16* __restrict__ dst,
                             int rows, int cols, int colsPad) {
    long i = (long)blockIdx.x * blockDim.x + threadIdx.x;
    long total = (long)rows * colsPad;
    if (i >= total) return;
    int r = (int)(i / colsPad);
    int c = (int)(i % colsPad);
    dst[i] = (c < cols) ? (bf16)src[(long)r * cols + c] : (bf16)0.0f;
}

// dst (Cpad x Rpad) row-major = transpose of src (R x Ccols), zero padded.
__global__ void transpose_cvt_bf16(const float* __restrict__ src, bf16* __restrict__ dst,
                                   int R, int Ccols, int Rpad, int Cpad) {
    long i = (long)blockIdx.x * blockDim.x + threadIdx.x;
    long total = (long)Cpad * Rpad;
    if (i >= total) return;
    int c = (int)(i / Rpad);
    int r = (int)(i % Rpad);
    dst[i] = (c < Ccols && r < R) ? (bf16)src[(long)r * Ccols + c] : (bf16)0.0f;
}

__global__ void zero_state(bf16* __restrict__ h, float* __restrict__ c, int n) {
    int i = blockIdx.x * blockDim.x + threadIdx.x;
    if (i < n) { h[i] = (bf16)0.0f; c[i] = 0.0f; }
}

// ---------------------------------------------------------------------------
// Stage 1: feat = tanh(BN(tanh(x @ W_fe + b_fe)))      (BT x H), bf16 out
// grid (BT/64, H/64), block 128; block tile 64x64, wave slice 16x64.
// ---------------------------------------------------------------------------
__global__ void featext_wmma(const bf16* __restrict__ Xb,   // (BT x FPAD)
                             const bf16* __restrict__ WfeT, // (H x FPAD)
                             const float* __restrict__ b_fe,
                             const float* __restrict__ g1, const float* __restrict__ be1,
                             const float* __restrict__ m1, const float* __restrict__ v1,
                             bf16* __restrict__ feat) {
    __shared__ bf16 lbs[64 * LDB_STR];
    const int wave = threadIdx.x >> 5;
    const int row0 = blockIdx.x * 64 + wave * 16;
    const int col0 = blockIdx.y * 64;
    v8f acc[4] = {{}, {}, {}, {}};
#pragma unroll 1
    for (int k0 = 0; k0 < FPAD; k0 += 32) {
        __syncthreads();                         // previous-iter reads done
        stage_b_tile(lbs, WfeT, FPAD, col0, k0);
        v16bf a = load_frag_a(Xb, FPAD, row0, k0);
        __builtin_prefetch(Xb + (long)row0 * FPAD + k0 + 32, 0, 1);
        __syncthreads();                         // tile visible
        v16bf b[4];
#pragma unroll
        for (int j = 0; j < 4; ++j) b[j] = load_frag_b_lds(lbs, j);
#pragma unroll
        for (int j = 0; j < 4; ++j) acc[j] = wmma_bf16(a, b[j], acc[j]);
    }
    const int lane = threadIdx.x & 31;
    const int n    = lane & 15;
    const int half = lane >> 4;
#pragma unroll
    for (int j = 0; j < 4; ++j) {
        const int col  = col0 + 16 * j + n;
        const float bias = b_fe[col];
        const float mu   = m1[col];
        const float sc   = rsqrtf(v1[col] + BN_EPS) * g1[col];
        const float bb   = be1[col];
#pragma unroll
        for (int r = 0; r < 8; ++r) {
            int row = row0 + half * 8 + r;
            float v = tanhf(acc[j][r] + bias);
            v = tanhf((v - mu) * sc + bb);
            feat[(long)row * HDIM + col] = (bf16)v;
        }
    }
}

// ---------------------------------------------------------------------------
// Stage 2a (per timestep): z = x_t @ kernel + h @ rec_kernel + bias (256x4096)
// grid (4, 64), block 128; block tile 64x64, wave slice 16x64.
// feat_t has row stride T*H (row = batch index).
// ---------------------------------------------------------------------------
__global__ void lstm_zgemm(const bf16* __restrict__ feat_t, // stride T*H
                           const bf16* __restrict__ h_prev, // (256 x H)
                           const bf16* __restrict__ kT,     // (4H x H)
                           const bf16* __restrict__ rT,     // (4H x H)
                           const float* __restrict__ bias,
                           float* __restrict__ z) {
    __shared__ bf16 lbs[64 * LDB_STR];
    const int wave = threadIdx.x >> 5;
    const int row0 = blockIdx.x * 64 + wave * 16;
    const int col0 = blockIdx.y * 64;
    v8f acc[4] = {{}, {}, {}, {}};
#pragma unroll 1
    for (int k0 = 0; k0 < HDIM; k0 += 32) {
        __syncthreads();
        stage_b_tile(lbs, kT, HDIM, col0, k0);
        v16bf a = load_frag_a(feat_t, (long)TSTEPS * HDIM, row0, k0);
        __builtin_prefetch(kT + (long)col0 * HDIM + k0 + 32, 0, 1);
        __syncthreads();
        v16bf b[4];
#pragma unroll
        for (int j = 0; j < 4; ++j) b[j] = load_frag_b_lds(lbs, j);
#pragma unroll
        for (int j = 0; j < 4; ++j) acc[j] = wmma_bf16(a, b[j], acc[j]);
    }
#pragma unroll 1
    for (int k0 = 0; k0 < HDIM; k0 += 32) {
        __syncthreads();
        stage_b_tile(lbs, rT, HDIM, col0, k0);
        v16bf a = load_frag_a(h_prev, HDIM, row0, k0);
        __builtin_prefetch(rT + (long)col0 * HDIM + k0 + 32, 0, 1);
        __syncthreads();
        v16bf b[4];
#pragma unroll
        for (int j = 0; j < 4; ++j) b[j] = load_frag_b_lds(lbs, j);
#pragma unroll
        for (int j = 0; j < 4; ++j) acc[j] = wmma_bf16(a, b[j], acc[j]);
    }
    const int lane = threadIdx.x & 31;
    const int n    = lane & 15;
    const int half = lane >> 4;
#pragma unroll
    for (int j = 0; j < 4; ++j) {
        const int col = col0 + 16 * j + n;
        const float bs = bias[col];
#pragma unroll
        for (int r = 0; r < 8; ++r)
            z[(long)(row0 + half * 8 + r) * H4 + col] = acc[j][r] + bs;
    }
}

// ---------------------------------------------------------------------------
// Stage 2b (per timestep): gates + state update (elementwise, 256*1024 threads)
// ---------------------------------------------------------------------------
__global__ void lstm_gates(const float* __restrict__ z,
                           const float* __restrict__ pi, const float* __restrict__ pf,
                           const float* __restrict__ po,
                           float* __restrict__ c, bf16* __restrict__ h,
                           bf16* __restrict__ h_seq, int t) {
    int i = blockIdx.x * blockDim.x + threadIdx.x;   // 0 .. 256*1024-1
    if (i >= BATCH * HDIM) return;
    int b = i >> 10;
    int j = i & (HDIM - 1);
    const float* zr = z + (long)b * H4;
    float zi = zr[j], zf = zr[j + HDIM], zc = zr[j + 2 * HDIM], zo = zr[j + 3 * HDIM];
    float co = c[i];
    float ig = sigm(zi + co * pi[j]);
    float fg = sigm(zf + co * pf[j]);
    float cn = fg * co + ig * tanhf(zc);
    float og = sigm(zo + cn * po[j]);
    float hn = og * tanhf(cn);
    c[i] = cn;
    h[i] = (bf16)hn;
    h_seq[((long)b * TSTEPS + t) * HDIM + j] = (bf16)hn;
}

// ---------------------------------------------------------------------------
// Stage 3: out = tanh(BN(h_seq)) @ W_out     (BT x 10), N padded to 16.
// BN+tanh applied on-the-fly to A fragments. grid 512, block 128.
// ---------------------------------------------------------------------------
__global__ void logits_wmma(const bf16* __restrict__ h_seq, // (BT x H)
                            const float* __restrict__ g2, const float* __restrict__ be2,
                            const float* __restrict__ m2, const float* __restrict__ v2,
                            const bf16* __restrict__ WoT,   // (16 x H), rows>=10 zero
                            float* __restrict__ out) {
    const int wave = threadIdx.x >> 5;
    const int row0 = (blockIdx.x * 4 + wave) * 16;
    const int lane = threadIdx.x & 31;
    const int mn   = lane & 15;
    const int half = lane >> 4;
    v8f acc = {};
#pragma unroll 1
    for (int k0 = 0; k0 < HDIM; k0 += 32) {
        const bf16* p0 = h_seq + (long)(row0 + mn) * HDIM + (k0 + half * 16);
        v8bf lo = *(const v8bf*)p0;
        v8bf hi = *(const v8bf*)(p0 + 8);
        v16bf a;
#pragma unroll
        for (int i = 0; i < 8; ++i) {
            int k = k0 + half * 16 + i;
            float x = (float)lo[i];
            a[i] = (bf16)tanhf((x - m2[k]) * rsqrtf(v2[k] + BN_EPS) * g2[k] + be2[k]);
        }
#pragma unroll
        for (int i = 0; i < 8; ++i) {
            int k = k0 + half * 16 + 8 + i;
            float x = (float)hi[i];
            a[8 + i] = (bf16)tanhf((x - m2[k]) * rsqrtf(v2[k] + BN_EPS) * g2[k] + be2[k]);
        }
        v16bf b = load_frag_b(WoT, HDIM, 0, k0);
        acc = wmma_bf16(a, b, acc);
    }
    if (mn < CDIM) {
#pragma unroll
        for (int r = 0; r < 8; ++r)
            out[(long)(row0 + half * 8 + r) * CDIM + mn] = acc[r];
    }
}

// ---------------------------------------------------------------------------
// Host side
// ---------------------------------------------------------------------------
extern "C" void kernel_launch(void* const* d_in, const int* in_sizes, int n_in,
                              void* d_out, int out_size, void* d_ws, size_t ws_size,
                              hipStream_t stream) {
    const float* x       = (const float*)d_in[0];
    const float* W_fe    = (const float*)d_in[1];
    const float* b_fe    = (const float*)d_in[2];
    const float* gamma1  = (const float*)d_in[3];
    const float* beta1   = (const float*)d_in[4];
    const float* mean1   = (const float*)d_in[5];
    const float* var1    = (const float*)d_in[6];
    const float* kernelW = (const float*)d_in[7];
    const float* recW    = (const float*)d_in[8];
    const float* bias    = (const float*)d_in[9];
    const float* peep_i  = (const float*)d_in[10];
    const float* peep_f  = (const float*)d_in[11];
    const float* peep_o  = (const float*)d_in[12];
    const float* gamma2  = (const float*)d_in[13];
    const float* beta2   = (const float*)d_in[14];
    const float* mean2   = (const float*)d_in[15];
    const float* var2    = (const float*)d_in[16];
    const float* W_out   = (const float*)d_in[17];
    float* out = (float*)d_out;

    // Workspace carve-up (all sizes already multiples of 256 bytes).
    char* ws = (char*)d_ws;
    size_t off = 0;
    bf16* xb    = (bf16*)(ws + off); off += (size_t)BT * FPAD * sizeof(bf16);     // 52.4 MB
    bf16* wfeT  = (bf16*)(ws + off); off += (size_t)HDIM * FPAD * sizeof(bf16);   // 1.6 MB
    bf16* kT    = (bf16*)(ws + off); off += (size_t)H4 * HDIM * sizeof(bf16);     // 8.4 MB
    bf16* rT    = (bf16*)(ws + off); off += (size_t)H4 * HDIM * sizeof(bf16);     // 8.4 MB
    bf16* woT   = (bf16*)(ws + off); off += (size_t)16 * HDIM * sizeof(bf16);     // 32 KB
    bf16* feat  = (bf16*)(ws + off); off += (size_t)BT * HDIM * sizeof(bf16);     // 67 MB
    bf16* hseq  = (bf16*)(ws + off); off += (size_t)BT * HDIM * sizeof(bf16);     // 67 MB
    float* zbuf = (float*)(ws + off); off += (size_t)BATCH * H4 * sizeof(float);  // 4.2 MB
    bf16* hbf   = (bf16*)(ws + off); off += (size_t)BATCH * HDIM * sizeof(bf16);  // 0.5 MB
    float* cst  = (float*)(ws + off); off += (size_t)BATCH * HDIM * sizeof(float);// 1.0 MB
    (void)ws_size; (void)n_in; (void)in_sizes; (void)out_size;

    // ---- layout conversions ----
    {
        long tot = (long)BT * FPAD;
        pad_cvt_bf16<<<dim3((unsigned)((tot + 255) / 256)), dim3(256), 0, stream>>>(
            x, xb, BT, FDIM, FPAD);
    }
    {
        long tot = (long)HDIM * FPAD;   // W_fe (784 x 1024) -> (1024 x 800)
        transpose_cvt_bf16<<<dim3((unsigned)((tot + 255) / 256)), dim3(256), 0, stream>>>(
            W_fe, wfeT, FDIM, HDIM, FPAD, HDIM);
    }
    {
        long tot = (long)H4 * HDIM;     // kernel (1024 x 4096) -> (4096 x 1024)
        transpose_cvt_bf16<<<dim3((unsigned)((tot + 255) / 256)), dim3(256), 0, stream>>>(
            kernelW, kT, HDIM, H4, HDIM, H4);
        transpose_cvt_bf16<<<dim3((unsigned)((tot + 255) / 256)), dim3(256), 0, stream>>>(
            recW, rT, HDIM, H4, HDIM, H4);
    }
    {
        long tot = (long)16 * HDIM;     // W_out (1024 x 10) -> (16 x 1024), padded
        transpose_cvt_bf16<<<dim3((unsigned)((tot + 255) / 256)), dim3(256), 0, stream>>>(
            W_out, woT, HDIM, CDIM, HDIM, 16);
    }
    zero_state<<<dim3(BATCH * HDIM / 256), dim3(256), 0, stream>>>(hbf, cst, BATCH * HDIM);

    // ---- stage 1: feature extraction GEMM (64x64 block tiles) ----
    featext_wmma<<<dim3(BT / 64, HDIM / 64), dim3(128), 0, stream>>>(
        xb, wfeT, b_fe, gamma1, beta1, mean1, var1, feat);

    // ---- stage 2: recurrence, T sequential (gemm + gates) launches ----
    for (int t = 0; t < TSTEPS; ++t) {
        const bf16* feat_t = feat + (long)t * HDIM;   // row stride T*H, row = batch
        lstm_zgemm<<<dim3(BATCH / 64, H4 / 64), dim3(128), 0, stream>>>(
            feat_t, hbf, kT, rT, bias, zbuf);
        lstm_gates<<<dim3(BATCH * HDIM / 256), dim3(256), 0, stream>>>(
            zbuf, peep_i, peep_f, peep_o, cst, hbf, hseq, t);
    }

    // ---- stage 3: BN -> tanh -> logits GEMM ----
    logits_wmma<<<dim3(BT / 16 / 4), dim3(128), 0, stream>>>(
        hseq, gamma2, beta2, mean2, var2, woT, out);
}
